// ExpanderLinear_23691039605303
// MI455X (gfx1250) — matrix operand order repaired and checked by reading, compiled, not run on hardware
//
#include <hip/hip_runtime.h>

// ---------------------------------------------------------------------------
// ExpanderLinear: out[N,OUT_F] = input[N,IN_F] @ (weight*mask)^T
// fp32 reference, computed via bf16x3 split-precision WMMA on gfx1250.
//
// Roofline: 137 GFLOP over ~288MB HBM traffic -> memory floor ~12.4us at
// 23.3 TB/s, so the matrix pipes must carry the FLOPs. fp32 WMMA (16x16x4)
// needs 8 issues per K=32; bf16x3 (hi*hi + hi*lo + lo*hi in f32 accum) needs
// only 3 with ~2^-16 relative error — fp32-class accuracy at bf16 rates.
// ---------------------------------------------------------------------------

#define IN_F  2048
#define OUT_F 2048
#define NROW  16384

typedef __attribute__((ext_vector_type(16))) __bf16 v16bf;
typedef __attribute__((ext_vector_type(8)))  __bf16 v8bf;
typedef __attribute__((ext_vector_type(4)))  __bf16 v4bf;
typedef __attribute__((ext_vector_type(8)))  float  v8f;
typedef __attribute__((ext_vector_type(4)))  float  v4f;

// ---------------------------------------------------------------------------
// Prep: split fp32 (optionally masked) into hi/lo bf16 planes.
//   hi = bf16_rne(x), lo = bf16_rne(x - hi)  =>  x ~= hi + lo
// ---------------------------------------------------------------------------
__global__ __launch_bounds__(256) void split_masked_weight(
    const float* __restrict__ w, const float* __restrict__ m,
    __bf16* __restrict__ hi, __bf16* __restrict__ lo, int n) {
  int i = (blockIdx.x * blockDim.x + threadIdx.x) * 4;
  if (i >= n) return;
  v4f wv = *(const v4f*)(w + i);
  v4f mv = *(const v4f*)(m + i);
  v4bf hv, lv;
#pragma unroll
  for (int j = 0; j < 4; ++j) {
    float f = wv[j] * mv[j];
    __bf16 h = (__bf16)f;
    hv[j] = h;
    lv[j] = (__bf16)(f - (float)h);
  }
  *(v4bf*)(hi + i) = hv;
  *(v4bf*)(lo + i) = lv;
}

__global__ __launch_bounds__(256) void split_input(
    const float* __restrict__ a,
    __bf16* __restrict__ hi, __bf16* __restrict__ lo, int n) {
  int i = (blockIdx.x * blockDim.x + threadIdx.x) * 4;
  if (i >= n) return;
  v4f av = *(const v4f*)(a + i);
  v4bf hv, lv;
#pragma unroll
  for (int j = 0; j < 4; ++j) {
    float f = av[j];
    __bf16 h = (__bf16)f;
    hv[j] = h;
    lv[j] = (__bf16)(f - (float)h);
  }
  *(v4bf*)(hi + i) = hv;
  *(v4bf*)(lo + i) = lv;
}

// ---------------------------------------------------------------------------
// Fragment loaders matching the CDNA5 16-bit WMMA VGPR layouts.
//
// A-matrix 16x32 (MxK), lane L: row m = L&15.
//   lanes 0-15:  VGPR0-3 hold K 0..7,  VGPR4-7 hold K 16..23
//   lanes 16-31: VGPR0-3 hold K 8..15, VGPR4-7 hold K 24..31
// => per lane: two contiguous 16B chunks at +kA and +kA+16 elems, kA={0,8}.
//
// B-matrix 32x16 (KxN), lane L: col n = L&15.
//   same K layout as A: lanes 0-15 hold K 0..7,16..23; lanes 16-31 hold K 8..15,24..31.
// Since B = Wm^T, column n of B is row n of Wm: K-contiguous in memory,
// so no transpose / LDS staging needed — two aligned b128 loads per frag.
// ---------------------------------------------------------------------------
__device__ inline v16bf load_frag_a(const __bf16* p) {
  union { v16bf v; v8bf h[2]; } u;
  u.h[0] = *(const v8bf*)(p);
  u.h[1] = *(const v8bf*)(p + 16);
  return u.v;
}

__device__ inline v16bf load_frag_b(const __bf16* p) {
  union { v16bf v; v8bf h[2]; } u;
  u.h[0] = *(const v8bf*)(p);
  u.h[1] = *(const v8bf*)(p + 16);
  return u.v;
}

__device__ inline v8f wmma_bf16(v16bf a, v16bf b, v8f c) {
  return __builtin_amdgcn_wmma_f32_16x16x32_bf16(
      /*neg_a=*/false, a, /*neg_b=*/false, b,
      /*c_mod=*/(short)0, c, /*reuse_a=*/false, /*reuse_b=*/false);
}

// ---------------------------------------------------------------------------
// GEMM: block = 256 threads = 8 waves in a 4(M) x 2(N) grid.
// Each wave computes a 32x64 C tile (2x4 WMMA tiles), block tile = 128x128.
// Per K=32 step and wave: 24x global_load_b128, 24x v_wmma_f32_16x16x32_bf16.
// ---------------------------------------------------------------------------
__global__ __launch_bounds__(256) void expander_gemm(
    const __bf16* __restrict__ Ahi, const __bf16* __restrict__ Alo,
    const __bf16* __restrict__ Whi, const __bf16* __restrict__ Wlo,
    float* __restrict__ out) {
  const int lane = threadIdx.x & 31;
  const int wave = threadIdx.x >> 5;
  const int wm = wave & 3;   // 0..3  (M direction)
  const int wn = wave >> 2;  // 0..1  (N direction)

  const int mb = blockIdx.y * 128 + wm * 32;   // wave M base (2 tiles of 16)
  const int nb = blockIdx.x * 128 + wn * 64;   // wave N base (4 tiles of 16)

  const int r15 = lane & 15;
  const int kA = (lane < 16) ? 0 : 8;   // A fragment chunk offset
  const int kB = (lane < 16) ? 0 : 8;   // B fragment chunk offset

  // Per-lane row base pointers (K-contiguous for both A and B=Wm^T).
  const __bf16* a_h = Ahi + (size_t)(mb + r15) * IN_F + kA;
  const __bf16* a_l = Alo + (size_t)(mb + r15) * IN_F + kA;
  const __bf16* b_h = Whi + (size_t)(nb + r15) * IN_F + kB;
  const __bf16* b_l = Wlo + (size_t)(nb + r15) * IN_F + kB;

  v8f acc[2][4];
#pragma unroll
  for (int i = 0; i < 2; ++i)
#pragma unroll
    for (int j = 0; j < 4; ++j) acc[i][j] = (v8f){};

  for (int kb = 0; kb < IN_F; kb += 32) {
    v16bf Ah[2], Al[2], Bh[4], Bl[4];
#pragma unroll
    for (int i = 0; i < 2; ++i) {
      Ah[i] = load_frag_a(a_h + (size_t)(i * 16) * IN_F + kb);
      Al[i] = load_frag_a(a_l + (size_t)(i * 16) * IN_F + kb);
    }
#pragma unroll
    for (int j = 0; j < 4; ++j) {
      Bh[j] = load_frag_b(b_h + (size_t)(j * 16) * IN_F + kb);
      Bl[j] = load_frag_b(b_l + (size_t)(j * 16) * IN_F + kb);
    }

    // bf16x3: hi*hi + hi*lo + lo*hi, f32 accumulate.
#pragma unroll
    for (int i = 0; i < 2; ++i) {
#pragma unroll
      for (int j = 0; j < 4; ++j) {
        acc[i][j] = wmma_bf16(Ah[i], Bh[j], acc[i][j]);
        acc[i][j] = wmma_bf16(Ah[i], Bl[j], acc[i][j]);
        acc[i][j] = wmma_bf16(Al[i], Bh[j], acc[i][j]);
      }
    }
  }

  // C/D layout: VGPR r, lanes 0-15 -> (M=r, N=lane); lanes 16-31 -> (M=8+r).
  const int n = lane & 15;
  const int mbase = (lane < 16) ? 0 : 8;
#pragma unroll
  for (int i = 0; i < 2; ++i) {
#pragma unroll
    for (int j = 0; j < 4; ++j) {
      union { v8f v; float f[8]; } u;
      u.v = acc[i][j];
#pragma unroll
      for (int r = 0; r < 8; ++r) {
        out[(size_t)(mb + i * 16 + mbase + r) * OUT_F + nb + j * 16 + n] = u.f[r];
      }
    }
  }
}

// ---------------------------------------------------------------------------
// Launch: prep (mask+split) -> GEMM. Workspace layout (bytes):
//   [0,        8MB)   Whi   (2048*2048 bf16)
//   [8MB,     16MB)   Wlo
//   [16MB,    80MB)   Ahi   (16384*2048 bf16)
//   [80MB,   144MB)   Alo
// ---------------------------------------------------------------------------
extern "C" void kernel_launch(void* const* d_in, const int* in_sizes, int n_in,
                              void* d_out, int out_size, void* d_ws, size_t ws_size,
                              hipStream_t stream) {
  const float* input  = (const float*)d_in[0];
  const float* weight = (const float*)d_in[1];
  const float* mask   = (const float*)d_in[2];
  float* out = (float*)d_out;

  char* ws = (char*)d_ws;
  const size_t WPLANE = (size_t)OUT_F * IN_F * sizeof(__bf16);  // 8 MB
  const size_t APLANE = (size_t)NROW * IN_F * sizeof(__bf16);   // 64 MB
  __bf16* Whi = (__bf16*)(ws);
  __bf16* Wlo = (__bf16*)(ws + WPLANE);
  __bf16* Ahi = (__bf16*)(ws + 2 * WPLANE);
  __bf16* Alo = (__bf16*)(ws + 2 * WPLANE + APLANE);

  const int wcount = OUT_F * IN_F;  // 4Mi elements
  split_masked_weight<<<wcount / 4 / 256, 256, 0, stream>>>(weight, mask, Whi, Wlo, wcount);

  const int acount = NROW * IN_F;   // 32Mi elements
  split_input<<<acount / 4 / 256, 256, 0, stream>>>(input, Ahi, Alo, acount);

  dim3 grid(OUT_F / 128, NROW / 128);  // (16, 128)
  expander_gemm<<<grid, 256, 0, stream>>>(Ahi, Alo, Whi, Wlo, out);
}
